// Feature_Generator_85839216378184
// MI455X (gfx1250) — matrix operand order repaired and checked by reading, not compiled, not run on hardware
//
#include <hip/hip_runtime.h>
#include <hip/hip_bf16.h>

// ---- problem constants (from reference) ----
#define NIN   128          // IN_F
#define NH    128          // H_F
#define SS    512          // S (nodes per subgraph)
#define NE    8192         // E (edges per subgraph)
#define NT    32           // T (remains)
#define NINJ  256          // INJ
#define GOUT  256          // 2*IN_F (gate output)
#define GHID  384          // 3*H_F  (gate hidden)

typedef __attribute__((ext_vector_type(16))) _Float16 v16h;
typedef __attribute__((ext_vector_type(8)))  _Float16 v8h;
typedef __attribute__((ext_vector_type(4)))  _Float16 v4h;
typedef __attribute__((ext_vector_type(8)))  float    v8f;

union AFrag { v16h v; v8h h[2]; };

// ---- LDS partition (bytes), total 273024 <= 320KB/WGP ----
#define OFF_X     0          // 512*128 f16  = 131072
#define OFF_AGG   131072     // 512*32  f32  =  65536
#define OFF_WF    196608     // 128*128 f16  =  32768 (B fragments)
#define OFF_EPK   229376     // 8192 u32     =  32768 (packed src|dst<<16)
#define OFF_SUB   262144     // 512 i32      =   2048
#define OFF_DINV  264192     // 512 f32      =   2048
#define OFF_DEG   266240     // 512 i32      =   2048
#define OFF_REM   268288     // 32  i32      =    128
#define OFF_RU    268416     // 128 f32      =    512
#define OFF_HID   268928     // 384 f32      =   1536
#define OFF_FD    270464     // 256 f32      =   1024
#define OFF_RED   271488     // 3*128 f32    =   1536
#define SMEM_TOTAL 273024

__device__ __forceinline__ float leaky(float x) { return x > 0.f ? x : 0.01f * x; }

__global__ void fg_zero_kernel(float* out) {
    if (threadIdx.x == 0 && blockIdx.x == 0) out[3 * NINJ * NIN] = 0.f;
}

__global__ __launch_bounds__(256, 1)
void fg_main_kernel(const float* __restrict__ feat,
                    const int*   __restrict__ ei,        // (INJ,2,E)
                    const int*   __restrict__ subset,    // (INJ,S)
                    const int*   __restrict__ remains,   // (INJ,T)
                    const int*   __restrict__ mapping,   // (INJ,)
                    const float* __restrict__ W1, const float* __restrict__ b1,
                    const float* __restrict__ W2, const float* __restrict__ b2,
                    const float* __restrict__ Wg, const float* __restrict__ bg,
                    const float* __restrict__ eps,
                    float* __restrict__ out) {
    extern __shared__ char smem[];
    _Float16* X    = (_Float16*)(smem + OFF_X);
    float*    AGG  = (float*)   (smem + OFF_AGG);
    _Float16* WF   = (_Float16*)(smem + OFF_WF);
    unsigned* EPK  = (unsigned*)(smem + OFF_EPK);
    int*      SUB  = (int*)     (smem + OFF_SUB);
    float*    DINV = (float*)   (smem + OFF_DINV);
    int*      DEG  = (int*)     (smem + OFF_DEG);
    int*      REM  = (int*)     (smem + OFF_REM);
    float*    RU   = (float*)   (smem + OFF_RU);
    float*    HID  = (float*)   (smem + OFF_HID);
    float*    FD   = (float*)   (smem + OFF_FD);
    float*    RED  = (float*)   (smem + OFF_RED);

    const int g = blockIdx.x;
    const int t = threadIdx.x;
    const int wave = t >> 5, lane = t & 31;
    const int lm = lane & 15, half = lane >> 4;

    const int* SRCI = ei + (size_t)g * 2 * NE;
    const int* DSTI = SRCI + NE;

    // ---- phase 1: indices ----
    for (int i = t; i < SS; i += 256) { SUB[i] = subset[g * SS + i]; DEG[i] = 0; }
    if (t < NT) REM[t] = remains[g * NT + t];
    __syncthreads();

    // ---- phase 2: ru_inj (f32 from global) + fs gather into f16 LDS ----
    if (t < NIN) {
        float s = 0.f;
        for (int r = 0; r < NT; ++r) {
            int node = SUB[REM[r]];
            s += feat[(size_t)node * NIN + t];
        }
        RU[t] = s * (1.f / NT);
    }
    {
        const float4* f4 = (const float4*)feat;
        for (int v = t; v < SS * (NIN / 4); v += 256) {
            int row = v >> 5, c4 = v & 31;
            float4 val = f4[(size_t)SUB[row] * (NIN / 4) + c4];
            v4h p; p[0] = (_Float16)val.x; p[1] = (_Float16)val.y;
                   p[2] = (_Float16)val.z; p[3] = (_Float16)val.w;
            *(v4h*)(&X[row * NIN + c4 * 4]) = p;
        }
    }
    // ---- phase 3: degrees (dst + self-loop) + packed edge cache ----
    for (int e = t; e < NE; e += 256) {
        int s = SRCI[e], d = DSTI[e];
        EPK[e] = (unsigned)s | ((unsigned)d << 16);
        atomicAdd(&DEG[d], 1);
    }
    __syncthreads();
    for (int i = t; i < SS; i += 256) DINV[i] = rsqrtf((float)(DEG[i] + 1));
    __syncthreads();

    // ---- GCN layers ----
    for (int layer = 0; layer < 2; ++layer) {
        const float* W = layer ? W2 : W1;
        const float* B = layer ? b2 : b1;

        // stage W in WMMA B-fragment order: WF[tile*512 + lane*16 + j] = W[k][n]
        for (int u = t; u < NH * NH; u += 256) {
            int j = u & 15, ln = (u >> 4) & 31, tile = u >> 9;
            int kt = tile & 3, nt = tile >> 2;
            int n = nt * 16 + (ln & 15);
            int k = kt * 32 + ((j >> 3) << 4) + ((ln >> 4) << 3) + (j & 7);
            WF[u] = (_Float16)W[k * NH + n];
        }
        __syncthreads();

        // in-place GEMM: X(512x128 f16) <- X @ W, f32 accumulate via WMMA.
        // Wave caches ALL its A fragments (4 mt x 4 kt) up front, so each
        // B-fragment load feeds 4 independent WMMAs and in-place writeback
        // of its own rows is safe.
        {
            AFrag a[4][4];
            #pragma unroll
            for (int mi = 0; mi < 4; ++mi) {
                const _Float16* xr = &X[((wave + 8 * mi) * 16 + lm) * NIN];
                #pragma unroll
                for (int kt = 0; kt < 4; ++kt) {
                    a[mi][kt].h[0] = *(const v8h*)(xr + kt * 32 + 8 * half);
                    a[mi][kt].h[1] = *(const v8h*)(xr + kt * 32 + 16 + 8 * half);
                }
            }
            #pragma unroll
            for (int nt = 0; nt < 8; ++nt) {
                v8f c[4];
                #pragma unroll
                for (int mi = 0; mi < 4; ++mi)
                    #pragma unroll
                    for (int r = 0; r < 8; ++r) c[mi][r] = 0.f;
                #pragma unroll
                for (int kt = 0; kt < 4; ++kt) {
                    v16h b = *(const v16h*)(&WF[((nt * 4 + kt) * 32 + lane) * 16]);
                    #pragma unroll
                    for (int mi = 0; mi < 4; ++mi)
                        c[mi] = __builtin_amdgcn_wmma_f32_16x16x32_f16(
                            false, a[mi][kt].v, false, b, (short)0, c[mi],
                            false, false);
                }
                #pragma unroll
                for (int mi = 0; mi < 4; ++mi) {
                    int mt = wave + 8 * mi;
                    #pragma unroll
                    for (int r = 0; r < 8; ++r)
                        X[(mt * 16 + r + 8 * half) * NIN + nt * 16 + lm] =
                            (_Float16)c[mi][r];
                }
            }
        }
        __syncthreads();

        // normalized aggregation, 4 feature chunks of 32 (in-place safe: chunk c
        // writes only cols [32c,32c+32), later chunks read higher cols of xw)
        for (int ch = 0; ch < 4; ++ch) {
            for (int u = t; u < SS * 32; u += 256) {        // self-loop init
                int row = u >> 5, f = u & 31;
                float di = DINV[row];
                AGG[u] = di * di * (float)X[row * NIN + ch * 32 + f];
            }
            __syncthreads();
            for (int e = wave; e < NE; e += 8) {            // edge scatter
                unsigned p = EPK[e];
                int s = (int)(p & 0xffffu), d = (int)(p >> 16);
                float nm = DINV[s] * DINV[d];
                float val = nm * (float)X[s * NIN + ch * 32 + lane];
                atomicAdd(&AGG[d * 32 + lane], val);
            }
            __syncthreads();
            for (int u = t; u < SS * 32; u += 256) {        // bias + leaky, f16 back
                int row = u >> 5, f = u & 31;
                float h = leaky(AGG[u] + B[ch * 32 + f]);
                X[row * NIN + ch * 32 + f] = (_Float16)h;
            }
            __syncthreads();
        }
    }

    // ---- pooling: hidden = [mean(h), h[mapping], mean(h[remains])] ----
    {   // node-mean: both thread halves sum 256 rows each
        int f = t & 127, hb = t >> 7;
        float sm = 0.f;
        for (int r = 0; r < 256; ++r) sm += (float)X[(hb * 256 + r) * NIN + f];
        RED[t] = sm;
    }
    __syncthreads();
    if (t < NH) {
        HID[t] = (RED[t] + RED[NH + t]) * (1.f / SS);
        int mp = mapping[g];
        HID[NH + t] = (float)X[mp * NIN + t];
        float ts = 0.f;
        for (int r = 0; r < NT; ++r) ts += (float)X[REM[r] * NIN + t];
        HID[2 * NH + t] = ts * (1.f / NT);
    }
    __syncthreads();

    // ---- gate: fd = leaky(hidden @ Wg + bg), one output per thread ----
    {
        float acc = bg[t];
        for (int k = 0; k < GHID; ++k) acc = fmaf(HID[k], Wg[k * GOUT + t], acc);
        FD[t] = leaky(acc);
    }
    __syncthreads();

    // ---- reparameterize + outputs + homophily terms ----
    float* out_feat = out;
    float* out_mu   = out + NINJ * NIN;
    float* out_sig  = out + 2 * NINJ * NIN;
    if (t < NIN) {
        float mu = FD[t];
        float sg = fabsf(FD[NIN + t]) + 1e-9f;
        float fv = mu + sg * eps[g * NIN + t];
        fv = fminf(fmaxf(fv, -1.f), 1.f);
        out_feat[g * NIN + t] = fv;
        out_mu[g * NIN + t]   = mu;
        out_sig[g * NIN + t]  = sg;
        float ru = RU[t];
        RED[t]           = ru * fv;
        RED[NIN + t]     = ru * ru;
        RED[2 * NIN + t] = fv * fv;
    }
    __syncthreads();
    if (t == 0) {
        float num = 0.f, ra = 0.f, fb = 0.f;
        for (int i = 0; i < NIN; ++i) {
            num += RED[i]; ra += RED[NIN + i]; fb += RED[2 * NIN + i];
        }
        float den = fmaxf(sqrtf(ra), 1e-8f) * fmaxf(sqrtf(fb), 1e-8f);
        atomicAdd(out + 3 * NINJ * NIN, (num / den) * (1.f / NINJ));
    }
}

extern "C" void kernel_launch(void* const* d_in, const int* in_sizes, int n_in,
                              void* d_out, int out_size, void* d_ws, size_t ws_size,
                              hipStream_t stream) {
    (void)in_sizes; (void)n_in; (void)out_size; (void)d_ws; (void)ws_size;
    const float* feat    = (const float*)d_in[0];
    const int*   ei      = (const int*)  d_in[1];
    const int*   subset  = (const int*)  d_in[2];
    const int*   remains = (const int*)  d_in[3];
    const int*   mapping = (const int*)  d_in[4];
    // d_in[5] = inj_num (unused)
    const float* W1 = (const float*)d_in[6];
    const float* b1 = (const float*)d_in[7];
    const float* W2 = (const float*)d_in[8];
    const float* b2 = (const float*)d_in[9];
    const float* Wg = (const float*)d_in[10];
    const float* bg = (const float*)d_in[11];
    const float* ep = (const float*)d_in[12];
    float* out = (float*)d_out;

    fg_zero_kernel<<<1, 32, 0, stream>>>(out);
    fg_main_kernel<<<dim3(NINJ), dim3(256), SMEM_TOTAL, stream>>>(
        feat, ei, subset, remains, mapping, W1, b1, W2, b2, Wg, bg, ep, out);
}
